// BitFeedForwardLegacy_17978733101753
// MI455X (gfx1250) — compile-verified, matrix operand order checked by reading, not run on hardware
//
#include <hip/hip_runtime.h>
#include <math.h>

#define EPS 1e-5f

typedef __attribute__((ext_vector_type(8))) int v8i;

// ---------------------------------------------------------------------------
// Stage 1a: per-block partial sums of |w| (deterministic two-stage reduction)
// ---------------------------------------------------------------------------
__global__ __launch_bounds__(256) void absum_partial_kernel(
    const float* __restrict__ w, size_t n4, float* __restrict__ partial) {
  // n4 = element count / 4; w is float4-aligned and n is a multiple of 4
  const float4* w4 = (const float4*)w;
  float s = 0.f;
  for (size_t i = (size_t)blockIdx.x * blockDim.x + threadIdx.x; i < n4;
       i += (size_t)gridDim.x * blockDim.x) {
    float4 v = w4[i];
    s += fabsf(v.x) + fabsf(v.y) + fabsf(v.z) + fabsf(v.w);
  }
  __shared__ float red[256];
  int tid = threadIdx.x;
  red[tid] = s;
  __syncthreads();
  for (int st = 128; st > 0; st >>= 1) {
    if (tid < st) red[tid] += red[tid + st];
    __syncthreads();
  }
  if (tid == 0) partial[blockIdx.x] = red[0];
}

__global__ __launch_bounds__(256) void absum_finalize_kernel(
    const float* __restrict__ partial, int nb, float* __restrict__ out) {
  float s = 0.f;
  for (int i = threadIdx.x; i < nb; i += blockDim.x) s += partial[i];
  __shared__ float red[256];
  int tid = threadIdx.x;
  red[tid] = s;
  __syncthreads();
  for (int st = 128; st > 0; st >>= 1) {
    if (tid < st) red[tid] += red[tid + st];
    __syncthreads();
  }
  if (tid == 0) *out = red[0];
}

// ---------------------------------------------------------------------------
// Stage 1b: ternary weight quantization (vectorized).
// scale = 1/max(mean|w|,EPS); q = clip(round(w*scale),-1,1) as int8.
// ---------------------------------------------------------------------------
__device__ __forceinline__ int pack_q8(float a, float b, float c, float d,
                                       float scale, float lo, float hi) {
  int b0 = (int)fminf(fmaxf(rintf(a * scale), lo), hi);
  int b1 = (int)fminf(fmaxf(rintf(b * scale), lo), hi);
  int b2 = (int)fminf(fmaxf(rintf(c * scale), lo), hi);
  int b3 = (int)fminf(fmaxf(rintf(d * scale), lo), hi);
  return (b0 & 0xff) | ((b1 & 0xff) << 8) | ((b2 & 0xff) << 16) | (b3 << 24);
}

__global__ __launch_bounds__(256) void wquant_kernel(
    const float* __restrict__ w, const float* __restrict__ sum, float invN,
    signed char* __restrict__ wq, size_t n4) {
  float mean  = fmaxf(*sum * invN, EPS);
  float scale = 1.f / mean;
  const float4* w4 = (const float4*)w;
  int* out4 = (int*)wq;
  for (size_t i = (size_t)blockIdx.x * blockDim.x + threadIdx.x; i < n4;
       i += (size_t)gridDim.x * blockDim.x) {
    float4 v = w4[i];
    out4[i] = pack_q8(v.x, v.y, v.z, v.w, scale, -1.f, 1.f);
  }
}

// ---------------------------------------------------------------------------
// Stage 2: per-token int8 absmax activation quantization (one block per row).
// ---------------------------------------------------------------------------
__global__ __launch_bounds__(256) void actquant_kernel(
    const float* __restrict__ x, signed char* __restrict__ xq,
    float* __restrict__ deq, int K) {
  __shared__ float red[256];
  const size_t row = blockIdx.x;
  const float4* xr = (const float4*)(x + row * (size_t)K);
  const int K4 = K >> 2;
  int tid = threadIdx.x;

  float m = 0.f;
  for (int i = tid; i < K4; i += blockDim.x) {
    float4 v = xr[i];
    m = fmaxf(m, fmaxf(fmaxf(fabsf(v.x), fabsf(v.y)),
                       fmaxf(fabsf(v.z), fabsf(v.w))));
  }
  red[tid] = m;
  __syncthreads();
  for (int st = 128; st > 0; st >>= 1) {
    if (tid < st) red[tid] = fmaxf(red[tid], red[tid + st]);
    __syncthreads();
  }
  float amax  = fmaxf(red[0], EPS);
  float scale = 127.f / amax;

  int* xqr = (int*)(xq + row * (size_t)K);
  for (int i = tid; i < K4; i += blockDim.x) {
    float4 v = xr[i];
    xqr[i] = pack_q8(v.x, v.y, v.z, v.w, scale, -128.f, 127.f);
  }
  if (tid == 0) deq[row] = amax * (1.f / 127.f);
}

// ---------------------------------------------------------------------------
// Stage 3: int8 x int8 -> int32 GEMM via V_WMMA_I32_16X16X64_IU8, software-
// pipelined (register double-buffered) so WMMAs overlap in-flight L2 loads.
//   out[m,n] = dot(A[m,:], B[n,:]) * aDeq[m] * wDeq + bias[n] (+ optional GELU)
// A: M x K row-major int8;  B: N x K row-major int8 (WMMA B operand = B^T).
// Block: 256 threads = 8 waves, arranged 4(M) x 2(N); block tile 128 x 128.
// Wave tile: 32(M) x 64(N) = 2 A-frags x 4 B-frags = 8 WMMAs per K-step.
// Fragment layout per ISA 7.12.2 (8-bit, 16x64): lane<16 holds K byte-chunks
// {0-7,16-23,32-39,48-55} of row (lane&15); lane>=16 holds the +8 chunks.
// ---------------------------------------------------------------------------
__device__ __forceinline__ v8i load_frag(const char* __restrict__ p, int k0) {
  union { v8i v; long l[4]; } f;
#pragma unroll
  for (int c = 0; c < 4; ++c)
    f.l[c] = *(const long*)(p + k0 + 16 * c);        // global_load_b64 x4
  return f.v;
}

__global__ __launch_bounds__(256) void gemm_iu8_kernel(
    const signed char* __restrict__ A, const signed char* __restrict__ B,
    const float* __restrict__ aDeq, const float* __restrict__ wSum, float wInvN,
    const float* __restrict__ bias, float* __restrict__ out,
    int M, int N, int K, int applyGelu) {
  const int tid    = threadIdx.x;
  const int wid    = tid >> 5;
  const int lane   = tid & 31;
  const int laneLo = lane & 15;
  const int laneHi = lane >> 4;

  const int tileM  = blockIdx.y * 128 + (wid >> 1) * 32;  // 4 M-groups of 32
  const int tileN0 = blockIdx.x * 128 + (wid & 1) * 64;   // 2 N-groups of 64

  const char* aP[2];
#pragma unroll
  for (int g = 0; g < 2; ++g)
    aP[g] = (const char*)A + (size_t)(tileM + 16 * g + laneLo) * K + laneHi * 8;
  const char* bP[4];
#pragma unroll
  for (int t = 0; t < 4; ++t)
    bP[t] = (const char*)B + (size_t)(tileN0 + 16 * t + laneLo) * K + laneHi * 8;

  v8i acc[8];
#pragma unroll
  for (int i = 0; i < 8; ++i) acc[i] = v8i{0};

  // Prologue: load k-step 0
  v8i aC[2], bC[4];
#pragma unroll
  for (int g = 0; g < 2; ++g) aC[g] = load_frag(aP[g], 0);
#pragma unroll
  for (int t = 0; t < 4; ++t) bC[t] = load_frag(bP[t], 0);

  // Main loop: issue next-step loads, then 8 independent WMMAs on current regs
  for (int k0 = 64; k0 < K; k0 += 64) {
    v8i aN[2], bN[4];
#pragma unroll
    for (int g = 0; g < 2; ++g) aN[g] = load_frag(aP[g], k0);
#pragma unroll
    for (int t = 0; t < 4; ++t) bN[t] = load_frag(bP[t], k0);

#pragma unroll
    for (int g = 0; g < 2; ++g)
#pragma unroll
      for (int t = 0; t < 4; ++t)
        acc[4 * g + t] = __builtin_amdgcn_wmma_i32_16x16x64_iu8(
            true, aC[g], true, bC[t], acc[4 * g + t], false, false);

#pragma unroll
    for (int g = 0; g < 2; ++g) aC[g] = aN[g];
#pragma unroll
    for (int t = 0; t < 4; ++t) bC[t] = bN[t];
  }

  // Drain: final k-step
#pragma unroll
  for (int g = 0; g < 2; ++g)
#pragma unroll
    for (int t = 0; t < 4; ++t)
      acc[4 * g + t] = __builtin_amdgcn_wmma_i32_16x16x64_iu8(
          true, aC[g], true, bC[t], acc[4 * g + t], false, false);

  // Epilogue: exact dequant + bias (+ exact erf-GELU for layer 1).
  const float wdeq = fmaxf(*wSum * wInvN, EPS);
#pragma unroll
  for (int g = 0; g < 2; ++g) {
    float ad[8];
#pragma unroll
    for (int r = 0; r < 8; ++r)
      ad[r] = aDeq[tileM + 16 * g + (laneHi << 3) + r] * wdeq;
#pragma unroll
    for (int t = 0; t < 4; ++t) {
      const int   col  = tileN0 + 16 * t + laneLo;     // n = lane & 15
      const float bcol = bias[col];
#pragma unroll
      for (int r = 0; r < 8; ++r) {                    // m = r + 8*(lane>=16)
        const int m = 16 * g + (laneHi << 3) + r;
        float v = (float)acc[4 * g + t][r] * ad[r] + bcol;
        if (applyGelu) v = 0.5f * v * (1.f + erff(v * 0.70710678118654752440f));
        out[(size_t)(tileM + m) * N + col] = v;
      }
    }
  }
}

// ---------------------------------------------------------------------------
// Host-side launch sequence
// ---------------------------------------------------------------------------
extern "C" void kernel_launch(void* const* d_in, const int* in_sizes, int n_in,
                              void* d_out, int out_size, void* d_ws, size_t ws_size,
                              hipStream_t stream) {
  (void)in_sizes; (void)n_in; (void)out_size; (void)ws_size;

  const float* x  = (const float*)d_in[0];   // (4,2048,2048)
  const float* w1 = (const float*)d_in[1];   // (8192,2048)
  const float* b1 = (const float*)d_in[2];   // (8192,)
  const float* w2 = (const float*)d_in[3];   // (2048,8192)
  const float* b2 = (const float*)d_in[4];   // (2048,)

  const int M = 4 * 2048;   // tokens
  const int D = 2048;       // model dim
  const int H = 8192;       // hidden dim
  const int NPART = 2048;   // partial-sum blocks per weight reduction

  // Workspace layout (all chunks 256B-aligned multiples)
  char* ws = (char*)d_ws;
  float* sums = (float*)ws;                                // [0]=sum|w1| [1]=sum|w2|
  size_t off = 256;
  float* part1 = (float*)(ws + off); off += (size_t)NPART * 4;
  float* part2 = (float*)(ws + off); off += (size_t)NPART * 4;
  signed char* xq  = (signed char*)(ws + off); off += (size_t)M * D;
  float*       xdq = (float*)(ws + off);       off += (size_t)M * 4;
  signed char* w1q = (signed char*)(ws + off); off += (size_t)H * D;
  signed char* w2q = (signed char*)(ws + off); off += (size_t)D * H;
  float*       h   = (float*)(ws + off);       off += (size_t)M * H * 4;
  signed char* hq  = (signed char*)(ws + off); off += (size_t)M * H;
  float*       hdq = (float*)(ws + off);       off += (size_t)M * 4;

  const size_t nW = (size_t)H * D;            // same count for both weights
  const float invNW = 1.f / (float)nW;

  // 1) per-tensor absmean of weights (deterministic two-stage)
  absum_partial_kernel<<<NPART, 256, 0, stream>>>(w1, nW / 4, part1);
  absum_partial_kernel<<<NPART, 256, 0, stream>>>(w2, nW / 4, part2);
  absum_finalize_kernel<<<1, 256, 0, stream>>>(part1, NPART, &sums[0]);
  absum_finalize_kernel<<<1, 256, 0, stream>>>(part2, NPART, &sums[1]);

  // 2) ternary-quantize weights
  wquant_kernel<<<4096, 256, 0, stream>>>(w1, &sums[0], invNW, w1q, nW / 4);
  wquant_kernel<<<4096, 256, 0, stream>>>(w2, &sums[1], invNW, w2q, nW / 4);

  // 3) quantize activations per token
  actquant_kernel<<<M, 256, 0, stream>>>(x, xq, xdq, D);

  // 4) GEMM1: (M x D)x(H x D)^T -> h, fused dequant+bias+exact GELU
  gemm_iu8_kernel<<<dim3(H / 128, M / 128), 256, 0, stream>>>(
      xq, w1q, xdq, &sums[0], invNW, b1, h, M, H, D, /*gelu=*/1);

  // 5) quantize h per token
  actquant_kernel<<<M, 256, 0, stream>>>(h, hq, hdq, H);

  // 6) GEMM2: (M x H)x(D x H)^T -> out, fused dequant+bias
  gemm_iu8_kernel<<<dim3(D / 128, M / 128), 256, 0, stream>>>(
      hq, w2q, hdq, &sums[1], invNW, b2, (float*)d_out, M, D, H, /*gelu=*/0);
}
